// HybridReducedQ_21938692948016
// MI455X (gfx1250) — matrix operand order repaired and checked by reading, not compiled, not run on hardware
//
#include <hip/hip_runtime.h>
#include <math.h>

// ---------------------------------------------------------------------------
// HybridReducedQ on MI455X (gfx1250, wave32, WMMA + TDM)
//
// q_out = cos(x_q1 + qp1) * cos(x_q2 + qp2)   (analytic collapse of circuit)
// out   = relu(relu([q_out, x_c] @ W1^T + b1) @ W2^T + b2) @ W3^T + b3
//
// Memory-bound: ~268 MB @ 23.3 TB/s ~= 11.5 us floor. Per 16-row tile only
// 5 v_wmma_f32_16x16x32_f16 against 4 KB of traffic -> WMMA throughput is
// never the limiter. This version uses the Tensor Data Mover to DMA each
// block's 256x63 f32 x_c tile into LDS (one descriptor, TENSORcnt-tracked),
// eliminating the entire VALU/VGPR staging loop; A-fragments are built by
// direct f32 LDS reads + in-register cvt to f16.
// ---------------------------------------------------------------------------

typedef _Float16 half16 __attribute__((ext_vector_type(16)));
typedef float    f32x8  __attribute__((ext_vector_type(8)));
typedef float    f32x4  __attribute__((ext_vector_type(4)));
typedef unsigned int uint32x4 __attribute__((ext_vector_type(4)));
typedef int          int32x4  __attribute__((ext_vector_type(4)));
typedef int          int32x8  __attribute__((ext_vector_type(8)));

#define ROWS      256          // rows per block
#define THREADS   256          // 8 waves (wave32)
#define NF        63           // classical features
#define SH_STRIDE 40           // halves per h1 row: 32 data + pad; 80B (16B-aligned, conflict-free)

__global__ __launch_bounds__(THREADS)
void hybrid_mlp_kernel(const float* __restrict__ xq,
                       const float* __restrict__ xc,
                       const float* __restrict__ qp,
                       const float* __restrict__ w1, const float* __restrict__ b1,
                       const float* __restrict__ w2, const float* __restrict__ b2,
                       const float* __restrict__ w3, const float* __restrict__ b3,
                       float* __restrict__ out, int Btot)
{
    __shared__ __align__(16) float    sF[ROWS * NF];             // f32 mirror of x_c tile (63 KB)
    __shared__ __align__(16) _Float16 sH[8][16 * SH_STRIDE];     // per-wave h1 [16][32] f16 (+pad)

    const int tid      = threadIdx.x;
    const int block0   = blockIdx.x * ROWS;
    const int rowsHere = min(ROWS, Btot - block0);

    // ---- Stage x_c f32 tile into LDS ----
#if defined(__gfx1250__) && __has_builtin(__builtin_amdgcn_tensor_load_to_lds)
    if (tid < 32) {    // wave 0 issues one TDM descriptor for the whole tile
        const unsigned ldsOff = (unsigned)(uintptr_t)(&sF[0]);   // flat LDS addr low 32b = WG-relative offset
        const unsigned long long ga =
            (unsigned long long)(uintptr_t)(xc + (size_t)block0 * NF);
        // D# group 0 (ISA 8.3): count=1, lds_addr, global_addr[56:0], type=2
        uint32x4 g0;
        g0[0] = 1u;                                              // count=1, is_restore=0, gather off
        g0[1] = ldsOff;                                          // lds_addr (bytes)
        g0[2] = (unsigned)(ga & 0xFFFFFFFFu);                    // global_addr[31:0]
        g0[3] = (unsigned)((ga >> 32) & 0x01FFFFFFu) | (2u << 30); // global_addr[56:32] | type=2
        // D# group 1 (ISA 8.4): data_size=4B; tensor 63 x rowsHere; tile 63 x rowsHere; dim0_stride=63
        int32x8 g1;
        g1[0] = (int)(2u << 16);                                 // data_size=2 (4 bytes), mask=0, no pad/iter
        g1[1] = (int)(((unsigned)NF & 0xFFFFu) << 16);           // tensor_dim0[15:0] = 63
        g1[2] = (int)(((unsigned)rowsHere & 0xFFFFu) << 16);     // tensor_dim1[15:0]
        g1[3] = (int)(((unsigned)NF) << 16);                     // tile_dim0 = 63 (tensor_dim1 hi16 = 0)
        g1[4] = (int)((unsigned)rowsHere & 0xFFFFu);             // tile_dim1 (tile_dim2 = 0 -> 2D)
        g1[5] = NF;                                              // tensor_dim0_stride[31:0] = 63
        g1[6] = 0;                                               // stride hi / dim1_stride lo (unused, 2D)
        g1[7] = 0;
        const int32x4 gz = {0, 0, 0, 0};                         // groups 2/3 unused (<=2D tensor)
#if __clang_major__ >= 23
        const int32x8 gz8 = {0, 0, 0, 0, 0, 0, 0, 0};
        __builtin_amdgcn_tensor_load_to_lds(g0, g1, gz, gz, gz8, 0);
#else
        __builtin_amdgcn_tensor_load_to_lds(g0, g1, gz, gz, 0);
#endif
        __builtin_amdgcn_s_wait_tensorcnt(0);                    // TDM done before barrier
    }
#else
    // Fallback: coalesced b128 copy global -> LDS
    if (rowsHere == ROWS) {
        const f32x4* src = (const f32x4*)(xc + (size_t)block0 * NF); // 16128 floats, 16B aligned
        f32x4* dst = (f32x4*)sF;
        for (int e4 = tid; e4 < (ROWS * NF) / 4; e4 += THREADS) dst[e4] = src[e4];
    } else {
        for (int e = tid; e < rowsHere * NF; e += THREADS)
            sF[e] = xc[(size_t)block0 * NF + e];
    }
#endif
    __syncthreads();

    // ---- wave-level WMMA pipeline ----
    const int lane = tid & 31;
    const int wid  = tid >> 5;
    const int ln   = lane & 15;     // row (A/D) or column (B) index
    const int kh   = lane >> 4;     // lane-group selects K slice

    const float qp1 = qp[1], qp2 = qp[2];

    // B-operand fragments (interleaved 8-elem K blocks, matching A: k = kh*8 + j, 16 + kh*8 + j-8)
    half16 bw1[2][2];               // [n-tile][k-step]  (W1: 32x64, row n is contiguous K)
    #pragma unroll
    for (int t = 0; t < 2; ++t) {
        #pragma unroll
        for (int s = 0; s < 2; ++s) {
            const float* wr = w1 + (size_t)(t * 16 + ln) * 64 + s * 32 + kh * 8;
            #pragma unroll
            for (int j = 0; j < 16; ++j) bw1[t][s][j] = (_Float16)wr[j + (j >= 8 ? 8 : 0)];
        }
    }
    half16 bw2;                     // W2: 16x32
    {
        const float* wr = w2 + (size_t)ln * 32 + kh * 8;
        #pragma unroll
        for (int j = 0; j < 16; ++j) bw2[j] = (_Float16)wr[j + (j >= 8 ? 8 : 0)];
    }
    const float bias1a = b1[ln];
    const float bias1b = b1[16 + ln];
    const float bias2v = b2[ln];
    const float w3v    = w3[ln];
    const float b3v    = b3[0];

    _Float16* hrow = &sH[wid][0];

    #pragma unroll
    for (int tt = 0; tt < 2; ++tt) {
        const int tileBase = wid * 32 + tt * 16;   // 16-row tile within block
        const int rowL = tileBase + ln;
        const int rowG = block0 + rowL;

        // per-row quantum expectation (analytic), computed by the lane that owns the row
        float q = 0.0f;
        if (rowG < Btot) {
            size_t g = (size_t)rowG * 3;
            q = cosf(xq[g + 1] + qp1) * cosf(xq[g + 2] + qp2);
        }

        // A fragments from f32 LDS tile (ISA 16-bit A-layout: interleaved 8-elem K blocks).
        // combined col c maps to: c==0 -> q ; c>0 -> sF[row*63 + c-1]
        const float* fRow = &sF[rowL * NF];
        half16 a0, a1;
        a0[0] = kh ? (_Float16)fRow[7] : (_Float16)q;            // c0 = kh*8
        #pragma unroll
        for (int j = 1; j < 8; ++j) a0[j] = (_Float16)fRow[kh * 8 + j - 1];
        #pragma unroll
        for (int j = 0; j < 8; ++j) {
            a0[8 + j] = (_Float16)fRow[16 + kh * 8 + j - 1];
            a1[j]     = (_Float16)fRow[32 + kh * 8 + j - 1];
            a1[8 + j] = (_Float16)fRow[48 + kh * 8 + j - 1];
        }

        // ---- layer 1: [16,64] x [64,32] = 2 n-tiles x 2 k-steps of WMMA ----
        #pragma unroll
        for (int t = 0; t < 2; ++t) {
            f32x8 acc = {0.f, 0.f, 0.f, 0.f, 0.f, 0.f, 0.f, 0.f};
            acc = __builtin_amdgcn_wmma_f32_16x16x32_f16(false, a0, false, bw1[t][0],
                                                         (short)0, acc, false, false);
            acc = __builtin_amdgcn_wmma_f32_16x16x32_f16(false, a1, false, bw1[t][1],
                                                         (short)0, acc, false, false);
            const float bias = t ? bias1b : bias1a;
            // D layout: lanes0-15 row=r, lanes16-31 row=r+8; col = ln + 16t
            #pragma unroll
            for (int r = 0; r < 8; ++r) {
                float h = fmaxf(acc[r] + bias, 0.0f);
                hrow[(r + 8 * kh) * SH_STRIDE + (ln + 16 * t)] = (_Float16)h;
            }
        }
        asm volatile("s_wait_dscnt 0x0" ::: "memory");   // intra-wave LDS RAW fence

        // re-layout h1 (D-layout) -> A fragment via LDS
        half16 a2;
        #pragma unroll
        for (int j = 0; j < 8; ++j) {
            a2[j]     = hrow[ln * SH_STRIDE + kh * 8 + j];
            a2[8 + j] = hrow[ln * SH_STRIDE + 16 + kh * 8 + j];
        }

        // ---- layer 2: [16,32] x [32,16] = 1 WMMA ----
        f32x8 acc2 = {0.f, 0.f, 0.f, 0.f, 0.f, 0.f, 0.f, 0.f};
        acc2 = __builtin_amdgcn_wmma_f32_16x16x32_f16(false, a2, false, bw2,
                                                      (short)0, acc2, false, false);

        // ---- layer 3: dot with w3 via 16-lane butterfly reduction ----
        float red[8];
        #pragma unroll
        for (int r = 0; r < 8; ++r) {
            float p = fmaxf(acc2[r] + bias2v, 0.0f) * w3v;
            p += __shfl_xor(p, 1, 16);
            p += __shfl_xor(p, 2, 16);
            p += __shfl_xor(p, 4, 16);
            p += __shfl_xor(p, 8, 16);
            red[r] = p + b3v;
        }
        if (ln == 0) {   // lane 0 -> rows 0-7, lane 16 -> rows 8-15
            const int gr0 = block0 + tileBase + 8 * kh;
            #pragma unroll
            for (int r = 0; r < 8; ++r)
                if (gr0 + r < Btot) out[gr0 + r] = red[r];
        }
    }
}

extern "C" void kernel_launch(void* const* d_in, const int* in_sizes, int n_in,
                              void* d_out, int out_size, void* d_ws, size_t ws_size,
                              hipStream_t stream)
{
    (void)n_in; (void)out_size; (void)d_ws; (void)ws_size;
    const float* xq = (const float*)d_in[0];
    const float* xc = (const float*)d_in[1];
    const float* qp = (const float*)d_in[2];
    const float* w1 = (const float*)d_in[3];
    const float* b1 = (const float*)d_in[4];
    const float* w2 = (const float*)d_in[5];
    const float* b2 = (const float*)d_in[6];
    const float* w3 = (const float*)d_in[7];
    const float* b3 = (const float*)d_in[8];
    float* out = (float*)d_out;

    const int B = in_sizes[0] / 3;                  // x_q is [B,3]
    const int grid = (B + ROWS - 1) / ROWS;
    hybrid_mlp_kernel<<<grid, THREADS, 0, stream>>>(xq, xc, qp, w1, b1, w2, b2,
                                                    w3, b3, out, B);
}
